// WienerSelfAttention_87720412053558
// MI455X (gfx1250) — compile-verified, matrix operand order checked by reading, not run on hardware
//
#include <hip/hip_runtime.h>
#include <hip/hip_bf16.h>

// ---------------------------------------------------------------------------
// Wiener self-attention for MI455X (gfx1250), wave32 + WMMA bf16 + async LDS.
// Fixed shapes from the reference: B=2, S=2048, D=1024, H=16, DK=64, gamma=0.1
//
// Math: scores[i,j] = sim(q_j, k_i)/8, probs = softmax_j(-scores).
//   -scores[i,j] = u_i . w_j - r_i/8; r_i is constant over j -> cancels in the
//   softmax. u_i = [-a_i/8, k_i*a_i/4], w_j = [q_j^2, q_j], a = 1/(k^2+gamma).
//   So logits are one K=128 bf16 GEMM.
//
// Roofline: mandatory 536 MB probs store ~= 23 us @ 23.3 TB/s dominates; all
// GEMMs (~77 GFLOP) ride bf16 WMMA and are free. Shared w-tiles are staged
// into LDS once per workgroup (4 waves reuse) via ASYNCcnt-tracked
// global_load_async_to_lds_b128, double-buffered against the WMMA stream.
// ---------------------------------------------------------------------------

typedef __attribute__((ext_vector_type(16))) __bf16 v16bf;
typedef __attribute__((ext_vector_type(8)))  __bf16 v8bf;
typedef __attribute__((ext_vector_type(8)))  float  v8f;
typedef int v4i_b128 __attribute__((vector_size(16)));   // matches builtin proto

union Frag { v16bf v; v8bf h[2]; };

#define NB 2
#define NS 2048
#define ND 1024
#define NH 16
#define NBH 32          // NB*NH
#define M_ROWS 4096     // NB*NS

#ifndef __has_builtin
#define __has_builtin(x) 0
#endif
#if __has_builtin(__builtin_amdgcn_global_load_async_to_lds_b128) && \
    __has_builtin(__builtin_amdgcn_s_wait_asynccnt)
#define HAVE_ASYNC_LDS 1
#else
#define HAVE_ASYNC_LDS 0
#endif

#define AS_GLOBAL __attribute__((address_space(1)))
#define AS_LOCAL  __attribute__((address_space(3)))

static __device__ __forceinline__ v8f wmma_bf16(const Frag& a, const Frag& b, v8f c) {
    return __builtin_amdgcn_wmma_f32_16x16x32_bf16(
        /*neg_a=*/false, a.v, /*neg_b=*/false, b.v,
        /*c_mod=*/(short)0, c, /*reuse_a=*/false, /*reuse_b=*/false);
}

// Load a 16x32 (rows x K) bf16 fragment from a row-major [rows, ldk] matrix.
// CDNA5 16-bit A/B^T layout: lane L (m=L&15, hi=L>>4) holds K = hi*8..+8 in
// VGPR0-3 and K = 16+hi*8..+8 in VGPR4-7 -> two contiguous 16B loads.
static __device__ __forceinline__ Frag load_frag(const __bf16* __restrict__ base,
                                                 int mlane, int hi, long ldk, int kbase) {
    Frag f;
    const __bf16* p = base + (long)mlane * ldk + kbase + hi * 8;
    f.h[0] = *(const v8bf*)(p);
    f.h[1] = *(const v8bf*)(p + 16);
    return f;
}

// Stage 32 rows x 128 cols of bf16 (8 KB) global -> LDS with all 128 threads.
static __device__ __forceinline__ void stage_w(const __bf16* __restrict__ gsrc,
                                               __bf16* __restrict__ lbuf, int tid) {
#if HAVE_ASYNC_LDS
#pragma unroll
    for (int c = 0; c < 4; c++) {
        int off = (tid + c * 128) * 16;
        __builtin_amdgcn_global_load_async_to_lds_b128(
            (AS_GLOBAL v4i_b128*)((const char*)gsrc + off),
            (AS_LOCAL  v4i_b128*)((char*)lbuf + off),
            /*offset=*/0, /*cpol=*/0);
    }
#else
#pragma unroll
    for (int c = 0; c < 4; c++) {
        int off = (tid + c * 128) * 16;
        *(v8bf*)((char*)lbuf + off) = *(const v8bf*)((const char*)gsrc + off);
    }
#endif
}

static __device__ __forceinline__ void stage_wait() {
#if HAVE_ASYNC_LDS
    __builtin_amdgcn_s_wait_asynccnt(0);
#endif
}

// ---------------------------------------------------------------------------
// Kernel 0: f32 -> bf16 conversion
// ---------------------------------------------------------------------------
__global__ void __launch_bounds__(256) cvt_bf16(const float* __restrict__ src,
                                               __bf16* __restrict__ dst, int n) {
    int i = blockIdx.x * 256 + threadIdx.x;
    if (i < n) dst[i] = (__bf16)src[i];
}

// ---------------------------------------------------------------------------
// Kernel 1: fused QKV projection (X @ W_z^T + b_z) with derived epilogues.
//   z=0 (q): w[bh,s,0:64]=q^2, w[bh,s,64:128]=q
//   z=1 (k): a=1/(k^2+0.1); u[bh,s,0:64]=-a/8, u[bh,s,64:128]=k*a/4
//   z=2 (v): vT[bh,d,s]=v
// Workgroup = 4 waves, tile 64(M) x 64(N); each wave: 16M x 64N.
// ---------------------------------------------------------------------------
__global__ void __launch_bounds__(128) qkv_gemm(
    const __bf16* __restrict__ Xbf,   // [4096,1024] row-major (m,k)
    const __bf16* __restrict__ Wbf,   // [3,1024,1024] row-major (n,k)
    const float*  __restrict__ bq, const float* __restrict__ bk,
    const float*  __restrict__ bv,
    __bf16* __restrict__ ubuf,        // [32,2048,128]
    __bf16* __restrict__ wbuf,        // [32,2048,128]
    __bf16* __restrict__ vT)          // [32,64,2048]
{
    const int lane  = threadIdx.x & 31;
    const int wave  = threadIdx.x >> 5;
    const int mlane = lane & 15;
    const int hi    = lane >> 4;
    const int z     = blockIdx.z;
    const int m0    = blockIdx.x * 64 + wave * 16;
    const int n0    = blockIdx.y * 64;

    const __bf16* Wz   = Wbf + (long)z * ND * ND;
    const float*  bias = (z == 0) ? bq : (z == 1 ? bk : bv);

    const __bf16* arow = Xbf + (long)(m0 + mlane) * ND;
    const __bf16* brow = Wz  + (long)(n0 + mlane) * ND;

    v8f acc[4] = {v8f{}, v8f{}, v8f{}, v8f{}};

    for (int k = 0; k < ND; k += 32) {
        __builtin_prefetch(arow + k + 128, 0, 3);
        __builtin_prefetch(brow + k + 128, 0, 3);
        Frag A;
        A.h[0] = *(const v8bf*)(arow + k + hi * 8);
        A.h[1] = *(const v8bf*)(arow + k + hi * 8 + 16);
#pragma unroll
        for (int t = 0; t < 4; t++) {
            Frag Bf;
            const __bf16* bp = brow + (long)t * 16 * ND + k + hi * 8;
            Bf.h[0] = *(const v8bf*)(bp);
            Bf.h[1] = *(const v8bf*)(bp + 16);
            acc[t] = wmma_bf16(A, Bf, acc[t]);
        }
    }

#pragma unroll
    for (int t = 0; t < 4; t++) {
#pragma unroll
        for (int r = 0; r < 8; r++) {
            int m = m0 + r + 8 * hi;           // global row (b*S+s)
            int n = n0 + t * 16 + mlane;       // global col (h*64+d)
            float val = acc[t][r] + bias[n];
            int b = m >> 11, s = m & 2047;
            int h = n >> 6,  d = n & 63;
            int bh = b * NH + h;
            if (z == 0) {
                long base = ((long)bh * NS + s) * 128;
                wbuf[base + d]      = (__bf16)(val * val);
                wbuf[base + 64 + d] = (__bf16)val;
            } else if (z == 1) {
                float a = 1.0f / (val * val + 0.1f);
                long base = ((long)bh * NS + s) * 128;
                ubuf[base + d]      = (__bf16)(-0.125f * a);
                ubuf[base + 64 + d] = (__bf16)(0.25f * val * a);
            } else {
                vT[((long)bh * 64 + d) * NS + s] = (__bf16)val;
            }
        }
    }
}

// ---------------------------------------------------------------------------
// Kernel 2: attention. 4 waves per workgroup share one (b,h); wave w owns the
// 16 i-rows [i0, i0+16). The shared w_j tiles (32 j-rows per stage) are
// async-staged global->LDS double-buffered; each wave reads its WMMA B-frags
// from LDS (4x reuse). Two-pass flash softmax; pass B recomputes logits
// (compute is free vs the 536 MB probs store), writes normalized probs, and
// accumulates P@V via WMMA with P re-staged through per-wave LDS.
// ---------------------------------------------------------------------------
__global__ void __launch_bounds__(128) wiener_attn(
    const __bf16* __restrict__ ubuf,   // [32,2048,128] rows i
    const __bf16* __restrict__ wbuf,   // [32,2048,128] rows j
    const __bf16* __restrict__ vT,     // [32,64,2048]
    float* __restrict__ ctx_out,       // [2,2048,1024]
    float* __restrict__ probs_out)     // [32,2048,2048]
{
    __shared__ __bf16 wtile[2][32 * 128];   // 16 KB double buffer (shared j tiles)
    __shared__ __bf16 pstash[4][16 * 32];   // 1 KB per wave (P restage for PV)

    const int tid   = threadIdx.x;
    const int lane  = tid & 31;
    const int wave  = tid >> 5;
    const int mlane = lane & 15;
    const int hi    = lane >> 4;
    const int bh    = blockIdx.x;                 // 0..31
    const int i0    = blockIdx.y * 64 + wave * 16;

    const __bf16* ubase = ubuf + ((long)bh * NS + i0) * 128;
    const __bf16* wmat  = wbuf + (long)bh * NS * 128;
    const __bf16* vbase = vT   + (long)bh * 64 * NS;
    float* probs = probs_out + (long)bh * NS * NS;

    // hoist the 4 A-fragments of u_i (K = 0,32,64,96)
    Frag U[4];
#pragma unroll
    for (int kk = 0; kk < 4; kk++) U[kk] = load_frag(ubase, mlane, hi, 128, kk * 32);

    // ---- pass A: online row max + sum(exp) over all j ----
    float pmax[8], psum[8];
#pragma unroll
    for (int r = 0; r < 8; r++) { pmax[r] = -__builtin_inff(); psum[r] = 0.0f; }

    stage_w(wmat, wtile[0], tid);
    stage_wait();
    __syncthreads();

    for (int s = 0; s < NS / 32; s++) {
        if (s + 1 < NS / 32)
            stage_w(wmat + (long)(s + 1) * 32 * 128, wtile[(s + 1) & 1], tid);
        const __bf16* wb = wtile[s & 1];
#pragma unroll
        for (int half = 0; half < 2; half++) {
            const __bf16* wrow = wb + half * 16 * 128;
            v8f acc = v8f{};
#pragma unroll
            for (int kk = 0; kk < 4; kk++) {
                Frag Bf = load_frag(wrow, mlane, hi, 128, kk * 32);  // ds_load_b128
                acc = wmma_bf16(U[kk], Bf, acc);
            }
#pragma unroll
            for (int r = 0; r < 8; r++) {
                float x  = acc[r];
                float nm = fmaxf(pmax[r], x);
                psum[r]  = psum[r] * __expf(pmax[r] - nm) + __expf(x - nm);
                pmax[r]  = nm;
            }
        }
        if (s + 1 < NS / 32) { stage_wait(); __syncthreads(); }
    }

    // reduce (max,sum) across the 16 lanes sharing each row (butterfly)
#pragma unroll
    for (int r = 0; r < 8; r++) {
#pragma unroll
        for (int off = 1; off < 16; off <<= 1) {
            float om = __shfl_xor(pmax[r], off, 32);
            float os = __shfl_xor(psum[r], off, 32);
            float nm = fmaxf(pmax[r], om);
            psum[r]  = psum[r] * __expf(pmax[r] - nm) + os * __expf(om - nm);
            pmax[r]  = nm;
        }
        psum[r] = 1.0f / psum[r];   // invsum
    }

    // ---- pass B: recompute logits, write probs, accumulate P@V ----
    __bf16* stash = pstash[wave];
    v8f cacc[4] = {v8f{}, v8f{}, v8f{}, v8f{}};

    __syncthreads();                 // everyone done with pass A buffers
    stage_w(wmat, wtile[0], tid);
    stage_wait();
    __syncthreads();

    for (int s = 0; s < NS / 32; s++) {
        if (s + 1 < NS / 32)
            stage_w(wmat + (long)(s + 1) * 32 * 128, wtile[(s + 1) & 1], tid);
        const __bf16* wb = wtile[s & 1];
#pragma unroll
        for (int half = 0; half < 2; half++) {
            const int j0 = (s * 2 + half) * 16;
            const __bf16* wrow = wb + half * 16 * 128;
            v8f acc = v8f{};
#pragma unroll
            for (int kk = 0; kk < 4; kk++) {
                Frag Bf = load_frag(wrow, mlane, hi, 128, kk * 32);
                acc = wmma_bf16(U[kk], Bf, acc);
            }
#pragma unroll
            for (int r = 0; r < 8; r++) {
                int   m = r + 8 * hi;
                float p = __expf(acc[r] - pmax[r]) * psum[r];   // normalized prob
                probs[(long)(i0 + m) * NS + j0 + mlane] = p;
                stash[m * 32 + half * 16 + mlane] = (__bf16)p;
            }
        }
        // P@V for the 32 staged j-columns (K=32 step)
        {
            Frag Pa;
            const __bf16* sp = stash + mlane * 32 + hi * 8;
            Pa.h[0] = *(const v8bf*)(sp);
            Pa.h[1] = *(const v8bf*)(sp + 16);
            const int kb = s * 32;
            __builtin_prefetch(vbase + kb + 32, 0, 3);
#pragma unroll
            for (int dt = 0; dt < 4; dt++) {
                Frag Vb = load_frag(vbase + (long)dt * 16 * NS, mlane, hi, NS, kb);
                cacc[dt] = wmma_bf16(Pa, Vb, cacc[dt]);
            }
        }
        if (s + 1 < NS / 32) { stage_wait(); __syncthreads(); }
    }

    // ---- store ctx: out[b, s, h*64+d] ----
    const int b = bh >> 4, h = bh & 15;
#pragma unroll
    for (int dt = 0; dt < 4; dt++) {
#pragma unroll
        for (int r = 0; r < 8; r++) {
            int s = i0 + r + 8 * hi;
            int d = dt * 16 + mlane;
            ctx_out[((long)b * NS + s) * ND + h * 64 + d] = cacc[dt][r];
        }
    }
}

// ---------------------------------------------------------------------------
// Launch. Workspace layout (bytes, all 16B aligned; total ~56.6 MB):
//   Xbf  @ 0          : 4096*1024 bf16   (8.39 MB)
//   Wbf  @ 8388608    : 3*1024*1024 bf16 (6.29 MB)
//   u    @ 14680064   : 32*2048*128 bf16 (16.78 MB)
//   w    @ 31457280   : 32*2048*128 bf16 (16.78 MB)
//   vT   @ 48234496   : 32*64*2048 bf16  (8.39 MB)
// d_out: ctx (4,194,304 f32) then probs (134,217,728 f32).
// ---------------------------------------------------------------------------
extern "C" void kernel_launch(void* const* d_in, const int* in_sizes, int n_in,
                              void* d_out, int out_size, void* d_ws, size_t ws_size,
                              hipStream_t stream) {
    const float* X  = (const float*)d_in[0];
    const float* Wq = (const float*)d_in[1];
    const float* bq = (const float*)d_in[2];
    const float* Wk = (const float*)d_in[3];
    const float* bk = (const float*)d_in[4];
    const float* Wv = (const float*)d_in[5];
    const float* bv = (const float*)d_in[6];

    char* ws = (char*)d_ws;
    __bf16* Xbf  = (__bf16*)(ws);
    __bf16* Wbf  = (__bf16*)(ws + 8388608);
    __bf16* ubuf = (__bf16*)(ws + 14680064);
    __bf16* wbuf = (__bf16*)(ws + 31457280);
    __bf16* vTb  = (__bf16*)(ws + 48234496);

    float* ctx   = (float*)d_out;
    float* probs = (float*)d_out + (long)NB * NS * ND;

    cvt_bf16<<<(M_ROWS * ND) / 256, 256, 0, stream>>>(X, Xbf, M_ROWS * ND);
    cvt_bf16<<<(ND * ND) / 256, 256, 0, stream>>>(Wq, Wbf, ND * ND);
    cvt_bf16<<<(ND * ND) / 256, 256, 0, stream>>>(Wk, Wbf + (long)ND * ND, ND * ND);
    cvt_bf16<<<(ND * ND) / 256, 256, 0, stream>>>(Wv, Wbf + 2l * ND * ND, ND * ND);

    qkv_gemm<<<dim3(M_ROWS / 64, ND / 64, 3), 128, 0, stream>>>(
        Xbf, Wbf, bq, bk, bv, ubuf, wbuf, vTb);

    wiener_attn<<<dim3(NBH, NS / 64), 128, 0, stream>>>(ubuf, wbuf, vTb, ctx, probs);
}
